// CausalMultiresConv1d_17987323035722
// MI455X (gfx1250) — matrix operand order repaired and checked by reading, not compiled, not run on hardware
//
#include <hip/hip_runtime.h>
#include <cstdint>
#include <cmath>

#define NC 64
#define NL 32768
#define NB 8
#define NDEPTH 8

typedef __attribute__((ext_vector_type(16))) __bf16 v16bf;
typedef __attribute__((ext_vector_type(8)))  float  v8f;

union ABregs { uint32_t u[8]; v16bf v; };

__device__ __forceinline__ unsigned short f2bf(float f) {
  uint32_t u = __builtin_bit_cast(uint32_t, f);
  u += 0x7fffu + ((u >> 16) & 1u);           // round-to-nearest-even
  return (unsigned short)(u >> 16);
}

__device__ __forceinline__ float gelu_exact(float v) {
  return 0.5f * v * (1.0f + erff(v * 0.70710678118654752f));
}

__global__ __launch_bounds__(256) void multires_wmma_kernel(
    const float* __restrict__ x, const float* __restrict__ h0,
    const float* __restrict__ h1, const float* __restrict__ w,
    float* __restrict__ out)
{
  // LDS: staged x (bf16, 8 batches x [T0-784, T0+2048)), reversed Toeplitz
  // filter images (2 parity copies), and f32 scratch for filter composition.
  __shared__ unsigned short xbf[NB][2832];
  __shared__ unsigned short Rt[2][864];
  __shared__ float fla[768];
  __shared__ float flb[768];
  __shared__ float Gacc[768];

  const int tid  = threadIdx.x;
  const int c    = blockIdx.x >> 4;     // channel
  const int slab = blockIdx.x & 15;     // 2048-wide t slab
  const int T0   = slab << 11;

  // ---------------- Phase 1: compose G[c] (length 766) ----------------
  float h0c[4], h1c[4];
  #pragma unroll
  for (int k = 0; k < 4; ++k) { h0c[k] = h0[c*4 + k]; h1c[k] = h1[c*4 + k]; }

  for (int i = tid; i < 768; i += 256) { fla[i] = (i == 0) ? 1.0f : 0.0f; Gacc[i] = 0.0f; }
  __syncthreads();

  float* cur = fla; float* nxt = flb;
  for (int lv = 0; lv < NDEPTH; ++lv) {
    const int   d  = 1 << lv;
    const float wi = w[c*10 + (NDEPTH - lv)];   // reference uses w[:, i], i = 8..1
    for (int i = tid; i < 768; i += 256) {
      float a0 = 0.0f, a1 = 0.0f;
      #pragma unroll
      for (int k = 0; k < 4; ++k) {
        const int t2 = i - d*k;
        if (t2 >= 0) { const float v = cur[t2]; a0 += h0c[k]*v; a1 += h1c[k]*v; }
      }
      nxt[i]   = a0;            // low-path filter for next level
      Gacc[i] += wi * a1;       // accumulate high-path term
    }
    __syncthreads();
    float* tmp = cur; cur = nxt; nxt = tmp;
  }
  {
    const float w0 = w[c*10 + 0];
    for (int i = tid; i < 768; i += 256) Gacc[i] += w0 * cur[i];
  }
  __syncthreads();

  // Reversed bf16 copies: Rt[par][idx] = G[(816+par) - idx], zero padded.
  // Descending Toeplitz pairs (G[a], G[a-1]) become aligned ascending b32 reads.
  for (int i = tid; i < 2*864; i += 256) {
    const int par = i / 864, idx = i % 864;
    const int tau = (816 + par) - idx;
    const float g = (tau >= 0 && tau < 768) ? Gacc[tau] : 0.0f;
    Rt[par][idx] = f2bf(g);
  }

  // ---------------- Phase 2: stage x -> bf16 LDS ----------------
  const int XU0 = T0 - 784;                       // left halo covers 766-tap filter
  for (int i = tid; i < NB * 708; i += 256) {     // 708 float4 per batch = 2832
    const int b  = i / 708, r4 = i % 708;
    const int u  = XU0 + r4*4;                    // multiple of 4; fully <0 or >=0
    float4 v = make_float4(0.f, 0.f, 0.f, 0.f);
    if (u >= 0) v = *(const float4*)(x + (size_t)(b*NC + c)*NL + u);
    const int o = r4*4;
    xbf[b][o+0] = f2bf(v.x); xbf[b][o+1] = f2bf(v.y);
    xbf[b][o+2] = f2bf(v.z); xbf[b][o+3] = f2bf(v.w);
  }
  __syncthreads();

  // ---------------- Phase 3: WMMA sweep ----------------
  const int wv   = tid >> 5, lane = tid & 31;
  const int W    = T0 + 256 * wv;                 // this wave covers [W, W+256)
  // A-operand lane constants (16-bit A 16x32 layout)
  const int mrow = lane & 15;
  const int par  = mrow & 1;
  const int kbA  = (lane < 16) ? 0 : 16;
  const int iA0  = (816 + par) - mrow + kbA;
  const unsigned short* Rrow = &Rt[par][0];
  // B-operand lane constants (16-bit B 32x16 layout); replicate batch for n>=8
  const int nn  = lane & 15;
  const int bb  = nn & 7;
  const int kbB = (lane < 16) ? 0 : 16;
  const unsigned short* xrow = &xbf[bb][0];
  const float wlast = w[c*10 + 9];

  for (int p = 0; p < 2; ++p) {                   // two 16-phases of the 32-stride tiling
    v8f D[8];
    #pragma unroll
    for (int i = 0; i < 8; ++i) { v8f z = {0,0,0,0,0,0,0,0}; D[i] = z; }
    ABregs Aring[8];
    const int ubase = 256*wv + 16*p;              // LDS index of u0 at j=0 (+784 halo folded in)

    #pragma unroll
    for (int j = 0; j < 32; ++j) {
      // One fresh Toeplitz A tile per chunk: d = j, o = 784 - 32*d (d in [0,24])
      if (j <= 24) {
        const int o = 784 - 32*j;
        const uint32_t* ap = (const uint32_t*)(Rrow + (iA0 - o));
        ABregs a;
        a.u[0] = ap[0]; a.u[1] = ap[1]; a.u[2] = ap[2];  a.u[3] = ap[3];
        a.u[4] = ap[4]; a.u[5] = ap[5]; a.u[6] = ap[6];  a.u[7] = ap[7];
        Aring[j & 7] = a;
      }
      // B tile: x[b, u0_j + k], k = kbB..kbB+15 packed pairs (contiguous 32B)
      const uint32_t* bp = (const uint32_t*)(xrow + (ubase + 32*j + kbB));
      ABregs bm;
      #pragma unroll
      for (int r = 0; r < 8; ++r) bm.u[r] = bp[r];
      // Up to 8 independent accumulations per chunk
      #pragma unroll
      for (int i = 0; i < 8; ++i) {
        const int dd = j - i;
        if (dd >= 0 && dd <= 24) {
          D[i] = __builtin_amdgcn_wmma_f32_16x16x32_bf16(
              false, Aring[dd & 7].v, false, bm.v, (short)0, D[i], false, false);
        }
      }
    }

    // Epilogue: D[m = t offset, n = batch]; lanes with n >= 8 carry replicas.
    if (nn < 8) {
      const int M0 = (lane < 16) ? 0 : 8;
      #pragma unroll
      for (int i = 0; i < 8; ++i) {
        const int t = W + 16*p + 32*i + M0;
        const size_t g = (size_t)(bb*NC + c)*NL + t;
        const float4 xa = *(const float4*)(x + g);
        const float4 xb = *(const float4*)(x + g + 4);
        float yv[8];
        #pragma unroll
        for (int r = 0; r < 8; ++r) yv[r] = D[i][r];
        yv[0] += wlast * xa.x; yv[1] += wlast * xa.y;
        yv[2] += wlast * xa.z; yv[3] += wlast * xa.w;
        yv[4] += wlast * xb.x; yv[5] += wlast * xb.y;
        yv[6] += wlast * xb.z; yv[7] += wlast * xb.w;
        float4 o0, o1;
        o0.x = gelu_exact(yv[0]); o0.y = gelu_exact(yv[1]);
        o0.z = gelu_exact(yv[2]); o0.w = gelu_exact(yv[3]);
        o1.x = gelu_exact(yv[4]); o1.y = gelu_exact(yv[5]);
        o1.z = gelu_exact(yv[6]); o1.w = gelu_exact(yv[7]);
        *(float4*)(out + g)     = o0;
        *(float4*)(out + g + 4) = o1;
      }
    }
  }
}

extern "C" void kernel_launch(void* const* d_in, const int* /*in_sizes*/, int /*n_in*/,
                              void* d_out, int /*out_size*/, void* /*d_ws*/, size_t /*ws_size*/,
                              hipStream_t stream) {
  const float* x  = (const float*)d_in[0];
  const float* h0 = (const float*)d_in[1];
  const float* h1 = (const float*)d_in[2];
  const float* w  = (const float*)d_in[3];
  float* out = (float*)d_out;
  // 64 channels x 16 slabs of 2048 t; 256 threads = 8 wave32, each wave owns 256 t.
  multires_wmma_kernel<<<dim3(NC * 16), dim3(256), 0, stream>>>(x, h0, h1, w, out);
}
